// BoxRPBAttention_76613626626661
// MI455X (gfx1250) — matrix operand order repaired and checked by reading, not compiled, not run on hardware
//
#include <hip/hip_runtime.h>
#include <math.h>

typedef _Float16 half_t;
typedef __attribute__((ext_vector_type(16))) _Float16 v16h;
typedef __attribute__((ext_vector_type(8)))  _Float16 v8h;
typedef __attribute__((ext_vector_type(8)))  float    v8f;
typedef __attribute__((ext_vector_type(4)))  float    v4f;

#define NQ 900
#define BB 4
#define CC 256
#define HH 8
#define HD 32
#define NN 4096
#define WW 64
#define RR 512
#define QSCALE 0.17677669529663687f /* 32^-0.5 */

// ---- workspace layout (bytes, all 256-aligned) ----
#define OFF_RX  ((size_t)0)          // 900*4*64*8 f32 = 7372800 B  [b][h][q][x]
#define OFF_RY  ((size_t)7372800)    // same
#define OFF_QH  ((size_t)14745600)   // 4*8*900*32  f16 = 1843200 B [b][h][q][d]
#define OFF_KH  ((size_t)16588800)   // 4*8*4096*32 f16 = 8388608 B
#define OFF_VH  ((size_t)24977408)   // 8388608 B
#define OFF_XH  ((size_t)33366016)   // 3600*256 f16 = 1843200 B
#define OFF_WQT ((size_t)35209216)   // 256*256 f16 = 131072 B each
#define OFF_WKT ((size_t)35340288)
#define OFF_WVT ((size_t)35471360)
#define OFF_WPT ((size_t)35602432)

union V16U { v16h v; v8h h[2]; };

__device__ __forceinline__ v16h concat8(v8h lo, v8h hi) {
    V16U u; u.h[0] = lo; u.h[1] = hi; return u.v;
}

__device__ __forceinline__ v8f wmma_f16(v16h a, v16h b, v8f c) {
    // D = A(16x32 f16) x B(32x16 f16) + C(16x16 f32)
    return __builtin_amdgcn_wmma_f32_16x16x32_f16(false, a, false, b, (short)0, c, false, false);
}

// LDS offset of a __shared__ object: flat generic address low 32 bits are the
// LDS byte address (ISA 10.2 aperture mapping).
__device__ __forceinline__ unsigned lds_off(const void* p) {
    return (unsigned)(uintptr_t)p;
}

// Async memory->LDS copy, 16B per lane, tracked by ASYNCcnt (no VGPR round trip).
__device__ __forceinline__ void async_b128(unsigned loff, const void* gaddr) {
    asm volatile("global_load_async_to_lds_b128 %0, %1, off"
                 :: "v"(loff), "v"((unsigned long long)(uintptr_t)gaddr)
                 : "memory");
}
__device__ __forceinline__ void wait_async0() {
    asm volatile("s_wait_asynccnt 0x0" ::: "memory");
}

// ---------------------------------------------------------------------------
// Kernel 1: W^T cast to f16:  WT[c][k] = (f16) W[k][c]   (256x256)
// ---------------------------------------------------------------------------
__global__ void wt_cast_kernel(const float* __restrict__ W, half_t* __restrict__ WT) {
    int t = blockIdx.x * 256 + threadIdx.x;     // 65536 threads
    int c = t >> 8, k = t & 255;
    WT[(c << 8) | k] = (half_t)W[(k << 8) | c];
}

// ---------------------------------------------------------------------------
// Kernel 2: RPE tables.  rxT/ryT layout [b][h][q][pos]  (pos = x or y)
// ---------------------------------------------------------------------------
__global__ void __launch_bounds__(256) rpe_kernel(
        const float* __restrict__ rp,
        const float* __restrict__ W1a, const float* __restrict__ b1a, const float* __restrict__ W2a,
        const float* __restrict__ W1b, const float* __restrict__ b1b, const float* __restrict__ W2b,
        float* __restrict__ rxT, float* __restrict__ ryT) {
    __shared__ float w10[RR], w11[RR], bb1[RR], w2[RR * HH];
    const int sel = blockIdx.y;                 // 0 = x branch, 1 = y branch
    const float* W1 = sel ? W1b : W1a;
    const float* b1 = sel ? b1b : b1a;
    const float* W2 = sel ? W2b : W2a;
    for (int i = threadIdx.x; i < RR; i += 256) { w10[i] = W1[i]; w11[i] = W1[RR + i]; bb1[i] = b1[i]; }
    for (int i = threadIdx.x; i < RR * HH; i += 256) w2[i] = W2[i];
    __syncthreads();

    int pid = blockIdx.x * 256 + threadIdx.x;   // < 900*4*64 = 230400
    int p = pid & 63; int t2 = pid >> 6; int b = t2 & 3; int q = t2 >> 2;
    const float* r4 = rp + (size_t)(q * BB + b) * 4;
    float cx = r4[0], cy = r4[1], sx = r4[2], sy = r4[3];
    const float SREF = 1024.0f;                 // w*stride = h*stride = 64*16
    float lo, hi;
    if (sel == 0) { lo = (cx - 0.5f * sx) * SREF; hi = (cx + 0.5f * sx) * SREF; }
    else          { lo = (cy - 0.5f * sy) * SREF; hi = (cy + 0.5f * sy) * SREF; }
    float pos = ((float)p + 0.5f) * 16.0f;
    float d0 = lo - pos, d1 = hi - pos;

    float acc[HH];
#pragma unroll
    for (int h2 = 0; h2 < HH; h2++) acc[h2] = 0.0f;
    for (int r = 0; r < RR; r++) {
        float hv = fmaf(d1, w11[r], fmaf(d0, w10[r], bb1[r]));
        hv = fmaxf(hv, 0.0f);
#pragma unroll
        for (int h2 = 0; h2 < HH; h2++) acc[h2] = fmaf(hv, w2[r * HH + h2], acc[h2]);
    }
    float* dst = sel ? ryT : rxT;
#pragma unroll
    for (int h2 = 0; h2 < HH; h2++)
        dst[((size_t)(b * HH + h2) * NQ + q) * WW + p] = acc[h2];
}

// ---------------------------------------------------------------------------
// Kernel 3: projection GEMM (f32 X @ f16 W^T + bias) * scale -> f16 [b][h][n][d]
//   rows r enumerate (n, b): r = n*B + b  (matches (seq,B,C) memory layout)
//   grid = (rows/16, 4), block = 128 (4 waves, one 16x16 tile per wave)
// ---------------------------------------------------------------------------
__global__ void __launch_bounds__(128) proj_qkv_kernel(
        const float* __restrict__ X, const half_t* __restrict__ WT,
        const float* __restrict__ bias, half_t* __restrict__ dst,
        int seqLen, float scale) {
    const int wv = threadIdx.x >> 5, lane = threadIdx.x & 31;
    const int l15 = lane & 15;
    const bool hiHalf = lane >= 16;
    const int tm = blockIdx.x;
    const int ct = blockIdx.y * 4 + wv;

    const int row = tm * 16 + l15;              // A-frag: lane holds row M=l15
    const float* xrow = X + (size_t)row * CC;
    const int wcol = ct * 16 + l15;             // B-frag: lane holds col N=l15
    const half_t* wrow = WT + (size_t)wcol * CC;

    v8f acc = {};
#pragma unroll
    for (int kk = 0; kk < 8; kk++) {            // K = 256 in chunks of 32
        const int ka = kk * 32 + (hiHalf ? 16 : 0);
        v4f x0 = *(const v4f*)(xrow + ka);
        v4f x1 = *(const v4f*)(xrow + ka + 4);
        v4f x2 = *(const v4f*)(xrow + ka + 8);
        v4f x3 = *(const v4f*)(xrow + ka + 12);
        v16h a;
#pragma unroll
        for (int i = 0; i < 4; i++) {
            a[i]      = (half_t)x0[i]; a[4 + i]  = (half_t)x1[i];
            a[8 + i]  = (half_t)x2[i]; a[12 + i] = (half_t)x3[i];
        }
        const int kb = kk * 32 + (hiHalf ? 16 : 0);
        v16h bf = *(const v16h*)(wrow + kb);
        acc = wmma_f16(a, bf, acc);
    }
    const float bcol = bias[wcol];
    const int hh = wcol >> 5, dd = wcol & 31;
#pragma unroll
    for (int j = 0; j < 8; j++) {               // C/D: VGPR j -> row j(+8), col l15
        int r = tm * 16 + j + (hiHalf ? 8 : 0);
        int n = r >> 2, b = r & 3;
        float v = (acc[j] + bcol) * scale;
        dst[((size_t)(b * HH + hh) * seqLen + n) * HD + dd] = (half_t)v;
    }
}

// ---------------------------------------------------------------------------
// Kernel 4: fused flash attention with box RPE.
//   grid = (ceil(NQ/128), H, B), block = 256 (8 waves; wave owns a 16-row q tile)
//   K tiles arrive via GLOBAL_LOAD_ASYNC_TO_LDS_B128 (ASYNCcnt), double-buffered.
// ---------------------------------------------------------------------------
__global__ void __launch_bounds__(256, 1) attn_kernel(
        const half_t* __restrict__ qh, const half_t* __restrict__ kh,
        const half_t* __restrict__ vh,
        const float* __restrict__ rxT, const float* __restrict__ ryT,
        half_t* __restrict__ xh) {
    __shared__ float  rxs[128][WW];             // 32 KB  rpe_x for block's q rows
    __shared__ float  rys[128][WW];             // 32 KB  rpe_y
    __shared__ half_t KtB[2][64][HD];           // 8 KB   key tiles   [k][d], dbl buf
    __shared__ half_t VtB[2][HD][64];           // 8 KB   value tiles [d][k] transposed
    __shared__ half_t Pl[8][16][64];            // 16 KB  per-wave P scratch

    const int b = blockIdx.z, h = blockIdx.y, q0 = blockIdx.x * 128;
    const int tid = threadIdx.x, wv = tid >> 5, lane = tid & 31;
    const int l15 = lane & 15;
    const bool hiHalf = lane >= 16;

    // async-stage RPE slices for this (b,h) and the block's 128 q rows
    const float* rxp = rxT + (size_t)(b * HH + h) * NQ * WW;
    const float* ryp = ryT + (size_t)(b * HH + h) * NQ * WW;
    for (int i = tid * 4; i < 128 * WW; i += 256 * 4) {   // 16B chunks within a row
        int m = i >> 6, x = i & 63;
        int q = q0 + m; if (q >= NQ) q = NQ - 1;
        async_b128(lds_off(&rxs[m][x]), rxp + (size_t)q * WW + x);
        async_b128(lds_off(&rys[m][x]), ryp + (size_t)q * WW + x);
    }

    // Q fragment (A layout): lane l15 -> row, d chunks 0..15/16..31
    const half_t* qbase = qh + (size_t)(b * HH + h) * NQ * HD;
    int qg = q0 + wv * 16 + l15; if (qg >= NQ) qg = NQ - 1;
    const int da = hiHalf ? 16 : 0;
    v16h qfrag = concat8(*(const v8h*)(qbase + (size_t)qg * HD + da),
                         *(const v8h*)(qbase + (size_t)qg * HD + da + 8));

    const half_t* kbase = kh + (size_t)(b * HH + h) * NN * HD;
    const half_t* vbase = vh + (size_t)(b * HH + h) * NN * HD;

    float mrow[8], lrow[8];
#pragma unroll
    for (int j = 0; j < 8; j++) { mrow[j] = -1e30f; lrow[j] = 0.0f; }
    v8f acc0 = {}, acc1 = {};

    // prologue: stage key/value tile 0
    const int key = tid >> 2;                   // 0..63
    const int dch = (tid & 3) * 8;              // 0,8,16,24
    {
        const size_t g0 = (size_t)key * HD + dch;
        async_b128(lds_off(&KtB[0][key][dch]), kbase + g0);
        v8h vv0 = *(const v8h*)(vbase + g0);
#pragma unroll
        for (int i = 0; i < 8; i++) VtB[0][dch + i][key] = vv0[i];
    }
    wait_async0();                              // rx/ry + K tile0 in LDS
    __syncthreads();

    for (int t = 0; t < NN / 64; t++) {         // 64 key tiles; tile == image row y=t
        const int cur = t & 1, nxt = cur ^ 1;

        // issue next tile's async K copy + V register load (overlaps compute)
        v8h vv = {};
        if (t + 1 < NN / 64) {
            const size_t g = (size_t)((t + 1) * 64 + key) * HD + dch;
            async_b128(lds_off(&KtB[nxt][key][dch]), kbase + g);
            vv = *(const v8h*)(vbase + g);
        }

        // scores: one WMMA per 16-key sub-tile (K dim = hd = 32)
        v8f sc[4];
#pragma unroll
        for (int nj = 0; nj < 4; nj++) {
            const int kk = nj * 16 + l15;       // B-frag: lane holds column (key)
            const int db = hiHalf ? 16 : 0;
            v16h bf = *(const v16h*)&KtB[cur][kk][db];
            v8f z = {};
            sc[nj] = wmma_f16(qfrag, bf, z);
        }
        // + rpe_x[q, x] + rpe_y[q, y=t]
#pragma unroll
        for (int nj = 0; nj < 4; nj++) {
            const int x = nj * 16 + l15;
#pragma unroll
            for (int j = 0; j < 8; j++) {
                const int m = wv * 16 + j + (hiHalf ? 8 : 0);
                sc[nj][j] += rxs[m][x] + rys[m][t];
            }
        }
        // per-row max of the 64-key tile (row lives on 16 lanes of a half-wave)
        float tmax[8];
#pragma unroll
        for (int j = 0; j < 8; j++)
            tmax[j] = fmaxf(fmaxf(sc[0][j], sc[1][j]), fmaxf(sc[2][j], sc[3][j]));
#pragma unroll
        for (int s = 1; s < 16; s <<= 1)
#pragma unroll
            for (int j = 0; j < 8; j++)
                tmax[j] = fmaxf(tmax[j], __shfl_xor(tmax[j], s, 16));

        float alpha[8], tsum[8];
#pragma unroll
        for (int j = 0; j < 8; j++) {
            float mn = fmaxf(mrow[j], tmax[j]);
            alpha[j] = __expf(mrow[j] - mn);
            mrow[j] = mn; tsum[j] = 0.0f;
        }
#pragma unroll
        for (int nj = 0; nj < 4; nj++)
#pragma unroll
            for (int j = 0; j < 8; j++) {
                float p = __expf(sc[nj][j] - mrow[j]);
                sc[nj][j] = p; tsum[j] += p;
            }
#pragma unroll
        for (int s = 1; s < 16; s <<= 1)
#pragma unroll
            for (int j = 0; j < 8; j++)
                tsum[j] += __shfl_xor(tsum[j], s, 16);
#pragma unroll
        for (int j = 0; j < 8; j++) {
            lrow[j] = lrow[j] * alpha[j] + tsum[j];
            acc0[j] *= alpha[j]; acc1[j] *= alpha[j];
        }

        // transpose P through wave-private LDS (C/D layout -> A layout)
#pragma unroll
        for (int nj = 0; nj < 4; nj++) {
            const int x = nj * 16 + l15;
#pragma unroll
            for (int j = 0; j < 8; j++)
                Pl[wv][j + (hiHalf ? 8 : 0)][x] = (half_t)sc[nj][j];
        }
        asm volatile("s_wait_dscnt 0x0" ::: "memory"); // intra-wave LDS RAW

        // PV: 2 key chunks of 32 x 2 hd chunks of 16 -> 4 WMMAs
#pragma unroll
        for (int kc = 0; kc < 2; kc++) {
            const int kbA = kc * 32 + (hiHalf ? 16 : 0);
            v16h af = concat8(*(const v8h*)&Pl[wv][l15][kbA],
                              *(const v8h*)&Pl[wv][l15][kbA + 8]);
            const int kbB = kc * 32 + (hiHalf ? 16 : 0);
            v16h bf0 = *(const v16h*)&VtB[cur][l15][kbB];        // d = 0..15
            acc0 = wmma_f16(af, bf0, acc0);
            v16h bf1 = *(const v16h*)&VtB[cur][16 + l15][kbB];   // d = 16..31
            acc1 = wmma_f16(af, bf1, acc1);
        }

        // finish the pipeline stage for tile t+1
        if (t + 1 < NN / 64) {
#pragma unroll
            for (int i = 0; i < 8; i++) VtB[nxt][dch + i][key] = vv[i];
        }
        wait_async0();                           // next K tile resident
        __syncthreads();                         // all waves done with cur + staged nxt
    }

    // normalize and store x as f16 rows [(q*B+b)][c], c = h*32 + d
    const int c0 = h * HD;
#pragma unroll
    for (int j = 0; j < 8; j++) {
        const int q = q0 + wv * 16 + j + (hiHalf ? 8 : 0);
        if (q < NQ) {
            const float inv = 1.0f / lrow[j];
            const size_t ro = ((size_t)q * BB + b) * CC;
            xh[ro + c0 + l15]      = (half_t)(acc0[j] * inv);
            xh[ro + c0 + 16 + l15] = (half_t)(acc1[j] * inv);
        }
    }
}

// ---------------------------------------------------------------------------
// Kernel 5: output projection, f16 X @ f16 Wp^T + bp -> f32 d_out (Nq,B,C)
// ---------------------------------------------------------------------------
__global__ void __launch_bounds__(128) proj_out_kernel(
        const half_t* __restrict__ Xh, const half_t* __restrict__ WT,
        const float* __restrict__ bias, float* __restrict__ out) {
    const int wv = threadIdx.x >> 5, lane = threadIdx.x & 31;
    const int l15 = lane & 15;
    const bool hiHalf = lane >= 16;
    const int tm = blockIdx.x;
    const int ct = blockIdx.y * 4 + wv;

    const int row = tm * 16 + l15;
    const half_t* xrow = Xh + (size_t)row * CC;
    const int wcol = ct * 16 + l15;
    const half_t* wrow = WT + (size_t)wcol * CC;

    v8f acc = {};
#pragma unroll
    for (int kk = 0; kk < 8; kk++) {
        const int ka = kk * 32 + (hiHalf ? 16 : 0);
        v16h a = concat8(*(const v8h*)(xrow + ka), *(const v8h*)(xrow + ka + 8));
        const int kb = kk * 32 + (hiHalf ? 16 : 0);
        v16h bf = *(const v16h*)(wrow + kb);
        acc = wmma_f16(a, bf, acc);
    }
    const float bcol = bias[wcol];
#pragma unroll
    for (int j = 0; j < 8; j++) {
        int r = tm * 16 + j + (hiHalf ? 8 : 0);
        out[(size_t)r * CC + wcol] = acc[j] + bcol;
    }
}

// ---------------------------------------------------------------------------
extern "C" void kernel_launch(void* const* d_in, const int* in_sizes, int n_in,
                              void* d_out, int out_size, void* d_ws, size_t ws_size,
                              hipStream_t stream) {
    (void)in_sizes; (void)n_in; (void)out_size; (void)ws_size;
    const float* query = (const float*)d_in[0];
    const float* rp    = (const float*)d_in[1];
    const float* kin   = (const float*)d_in[2];
    const float* vin   = (const float*)d_in[3];
    const float* Wq = (const float*)d_in[4];  const float* bq = (const float*)d_in[5];
    const float* Wk = (const float*)d_in[6];  const float* bk = (const float*)d_in[7];
    const float* Wv = (const float*)d_in[8];  const float* bv = (const float*)d_in[9];
    const float* W1a = (const float*)d_in[10]; const float* b1a = (const float*)d_in[11];
    const float* W2a = (const float*)d_in[12];
    const float* W1b = (const float*)d_in[13]; const float* b1b = (const float*)d_in[14];
    const float* W2b = (const float*)d_in[15];
    const float* Wp = (const float*)d_in[16]; const float* bp = (const float*)d_in[17];
    float* out = (float*)d_out;

    char* ws = (char*)d_ws;
    float*  rxT = (float*)(ws + OFF_RX);
    float*  ryT = (float*)(ws + OFF_RY);
    half_t* qhb = (half_t*)(ws + OFF_QH);
    half_t* khb = (half_t*)(ws + OFF_KH);
    half_t* vhb = (half_t*)(ws + OFF_VH);
    half_t* xhb = (half_t*)(ws + OFF_XH);
    half_t* WqT = (half_t*)(ws + OFF_WQT);
    half_t* WkT = (half_t*)(ws + OFF_WKT);
    half_t* WvT = (half_t*)(ws + OFF_WVT);
    half_t* WpT = (half_t*)(ws + OFF_WPT);

    wt_cast_kernel<<<256, 256, 0, stream>>>(Wq, WqT);
    wt_cast_kernel<<<256, 256, 0, stream>>>(Wk, WkT);
    wt_cast_kernel<<<256, 256, 0, stream>>>(Wv, WvT);
    wt_cast_kernel<<<256, 256, 0, stream>>>(Wp, WpT);

    rpe_kernel<<<dim3(900, 2), 256, 0, stream>>>(rp, W1a, b1a, W2a, W1b, b1b, W2b, rxT, ryT);

    proj_qkv_kernel<<<dim3(225, 4),  128, 0, stream>>>(query, WqT, bq, qhb, NQ, QSCALE);
    proj_qkv_kernel<<<dim3(1024, 4), 128, 0, stream>>>(kin,   WkT, bk, khb, NN, 1.0f);
    proj_qkv_kernel<<<dim3(1024, 4), 128, 0, stream>>>(vin,   WvT, bv, vhb, NN, 1.0f);

    attn_kernel<<<dim3(8, HH, BB), 256, 0, stream>>>(qhb, khb, vhb, rxT, ryT, xhb);

    proj_out_kernel<<<dim3(225, 4), 128, 0, stream>>>(xhb, WpT, bp, out);
}
